// SiameseNetWithLongformer_12670153523872
// MI455X (gfx1250) — matrix operand order repaired and checked by reading, not compiled, not run on hardware
//
#include <hip/hip_runtime.h>
#include <hip/hip_bf16.h>
#include <math.h>

// ---------------------------------------------------------------------------
// Shapes (fixed by the reference)
// ---------------------------------------------------------------------------
#define BB   2
#define SS   2048
#define DD   768
#define HH   12
#define DHD  64
#define LL   12
#define FF   3072
#define WW   256
#define CC   (SS / WW)          // 8 chunks
#define TOK  (BB * SS)          // 4096 rows

typedef __attribute__((ext_vector_type(16))) _Float16 v16h;
typedef __attribute__((ext_vector_type(8)))  _Float16 v8h;
typedef __attribute__((ext_vector_type(8)))  float    v8f;
typedef __attribute__((address_space(3)))    _Float16 lds_f16;

// ---------------------------------------------------------------------------
// Helpers
// ---------------------------------------------------------------------------
__device__ __forceinline__ float block_sum_256(float v, float* red) {
    int tid = threadIdx.x;
    red[tid] = v;
    __syncthreads();
    #pragma unroll
    for (int off = 128; off > 0; off >>= 1) {
        if (tid < off) red[tid] += red[tid + off];
        __syncthreads();
    }
    float r = red[0];
    __syncthreads();
    return r;
}

__device__ __forceinline__ v8f wmma_f16(v16h a, v16h b, v8f c) {
    return __builtin_amdgcn_wmma_f32_16x16x32_f16(false, a, false, b,
                                                  (short)0, c, false, false);
}

// Async global -> LDS copy, 16 bytes per lane, tracked by ASYNCcnt.
// vdst = LDS byte offset (per-lane), vaddr = 64-bit global address (per-lane).
__device__ __forceinline__ void async_copy_b128(unsigned lds_off,
                                                const _Float16* gaddr) {
    asm volatile("global_load_async_to_lds_b128 %0, %1, off"
                 :
                 : "v"(lds_off), "v"((unsigned long long)(size_t)gaddr)
                 : "memory");
}
__device__ __forceinline__ void wait_async_le1() {
    asm volatile("s_wait_asynccnt 0x1" ::: "memory");
}
__device__ __forceinline__ void wait_async_le0() {
    asm volatile("s_wait_asynccnt 0x0" ::: "memory");
}

// ---------------------------------------------------------------------------
// Weight packing: fp32 W[K,N] row-major -> f16 B-fragment blocked layout.
// Tile (kt,nt) occupies 512 halves: lane = (n%16) + (k%32 >= 16 ? 16 : 0),
// element e = k%16.  One v16h per lane per tile = contiguous 32B.
// ---------------------------------------------------------------------------
__global__ __launch_bounds__(256) void pack_w_kernel(const float* __restrict__ w,
                                                     _Float16* __restrict__ wp,
                                                     int K, int N) {
    int idx = blockIdx.x * 256 + threadIdx.x;
    if (idx >= K * N) return;
    int k = idx / N, n = idx % N;
    int kt = k >> 5, kk = k & 31, nt = n >> 4, nn = n & 15;
    int lane = nn + (kk & 16);
    int e = kk & 15;
    wp[(((size_t)kt * (N >> 4) + nt) * 32 + lane) * 16 + e] = (_Float16)w[idx];
}

// ---------------------------------------------------------------------------
// WMMA GEMM: C[M=4096, N] = A[M,K](f16) * Wpacked[K,N] + bias, with epilogues.
//   MODE 0: out f16 scattered to [B,H,S,DH]  (Q / K), value scaled by `scale`
//   MODE 1: out f32 row-major += residual    (O-proj / FFN2)
//   MODE 2: out f16 row-major, exact GELU    (FFN1)
//   MODE 3: out f16 scattered to [B,H,DH,S]  (V, transposed for P x V)
// Block = 256 threads (8 waves); wave w -> rows [blk.x*128 + 16w, +16),
// cols [blk.y*64, +64) as 4 WMMA tiles.
// The 4-tile (4 KB) packed-B K-slice is shared by all 8 waves: it is staged
// into LDS with double-buffered async global->LDS DMA (ASYNCcnt), so each
// slice is fetched once per block and the next DMA overlaps current WMMAs.
// ---------------------------------------------------------------------------
template <int MODE>
__global__ __launch_bounds__(256) void gemm_wmma_kernel(
    const _Float16* __restrict__ A, const _Float16* __restrict__ Wp,
    const float* __restrict__ bias, float* __restrict__ outF,
    _Float16* __restrict__ outH, const float* __restrict__ residual,
    int N, int K, float scale)
{
    __shared__ _Float16 sB[2 * 2048];   // two 4 KB B slices

    const int tid = threadIdx.x;
    const int lane = tid & 31;
    const int wave = tid >> 5;
    const int rowBase = blockIdx.x * 128 + wave * 16;
    const int colBase = blockIdx.y * 64;
    const int l15 = lane & 15;
    const int hi16 = lane & 16;          // 0 or 16
    const int o0 = hi16 ? 16 : 0;        // A-fragment half-group offset
    const int ntiles = N >> 4;
    const int colTile = colBase >> 4;

    const unsigned sBoff =
        (unsigned)(unsigned long long)(lds_f16*)&sB[0];   // LDS byte offset

    v8f acc[4] = {};

    // Prologue: stage K-slice 0 into buffer 0 (each thread copies 16 bytes).
    {
        const _Float16* g = Wp + ((size_t)0 * ntiles + colTile) * 512 + tid * 8;
        async_copy_b128(sBoff + tid * 16, g);
    }

    const int nk = K >> 5;
    for (int ik = 0; ik < nk; ++ik) {
        const int kb = ik << 5;
        if (ik + 1 < nk) {
            const _Float16* g =
                Wp + ((size_t)(ik + 1) * ntiles + colTile) * 512 + tid * 8;
            async_copy_b128(sBoff + ((ik + 1) & 1) * 4096 + tid * 16, g);
            wait_async_le1();            // current slice landed, next in flight
        } else {
            wait_async_le0();
        }
        __syncthreads();                 // all waves' DMA portions visible

        const _Float16* arow = A + (size_t)(rowBase + l15) * K + kb + o0;
        __builtin_prefetch(arow + 32, 0, 1);   // next K-step's A row
        v8h lo = *(const v8h*)(arow);
        v8h hi = *(const v8h*)(arow + 8);
        v16h a = __builtin_shufflevector(lo, hi, 0, 1, 2, 3, 4, 5, 6, 7,
                                         8, 9, 10, 11, 12, 13, 14, 15);

        const _Float16* bbase = sB + (ik & 1) * 2048 + lane * 16;
        #pragma unroll
        for (int nt = 0; nt < 4; ++nt) {
            v16h bf = *(const v16h*)(bbase + nt * 512);
            acc[nt] = wmma_f16(a, bf, acc[nt]);
        }
        __syncthreads();                 // reads done before buffer reuse
    }

    const int mOff = hi16 ? 8 : 0;
    #pragma unroll
    for (int nt = 0; nt < 4; ++nt) {
        int n = colBase + nt * 16 + l15;
        float bv = bias[n];
        if (MODE == 0) {
            int h = n >> 6, dh = n & 63;
            #pragma unroll
            for (int r = 0; r < 8; ++r) {
                int m = rowBase + mOff + r;
                int b = m >> 11, s = m & (SS - 1);
                float val = (acc[nt][r] + bv) * scale;
                outH[(((size_t)(b * HH + h)) * SS + s) * DHD + dh] = (_Float16)val;
            }
        } else if (MODE == 1) {
            #pragma unroll
            for (int r = 0; r < 8; ++r) {
                int m = rowBase + mOff + r;
                size_t idx = (size_t)m * N + n;
                outF[idx] = acc[nt][r] + bv + residual[idx];
            }
        } else if (MODE == 2) {
            #pragma unroll
            for (int r = 0; r < 8; ++r) {
                int m = rowBase + mOff + r;
                float xx = acc[nt][r] + bv;
                float g = 0.5f * xx * (1.0f + erff(xx * 0.70710678118f));
                outH[(size_t)m * N + n] = (_Float16)g;
            }
        } else {   // MODE 3: V transposed to [B,H,DH,S] (contiguous over r)
            int h = n >> 6, dh = n & 63;
            #pragma unroll
            for (int r = 0; r < 8; ++r) {
                int m = rowBase + mOff + r;
                int b = m >> 11, s = m & (SS - 1);
                float val = acc[nt][r] + bv;
                outH[(((size_t)(b * HH + h)) * DHD + dh) * SS + s] = (_Float16)val;
            }
        }
    }
}

// ---------------------------------------------------------------------------
// Sliding-window attention. One block (128 threads = 4 waves) handles one
// (b, h, chunk, 16-query-tile). q is pre-scaled by 1/sqrt(DH).
// Scores: S[16, 3W=768] = Q[16,64] x K^T  (WMMA, K-fragment = contiguous dh)
// Softmax in LDS, then Out[16,64] = P[16,768] x V[768,64] (WMMA); V is stored
// [B,H,DH,S] so each B fragment is one contiguous 32-byte load per lane.
// ---------------------------------------------------------------------------
__global__ __launch_bounds__(128) void attn_kernel(
    const _Float16* __restrict__ q, const _Float16* __restrict__ k,
    const _Float16* __restrict__ v, const int* __restrict__ mask,
    _Float16* __restrict__ aOut)
{
    __shared__ float sS[16 * 3 * WW];   // 48 KB

    int idx = blockIdx.x;
    int qt = idx & 15;
    int c  = (idx >> 4) & (CC - 1);
    int h  = (idx >> 7) % HH;
    int b  = idx / (16 * CC * HH);

    int tid = threadIdx.x;
    int lane = tid & 31;
    int wave = tid >> 5;
    int l15 = lane & 15;
    int hi16 = lane & 16;
    int o0 = hi16 ? 16 : 0;
    size_t headBase = ((size_t)(b * HH + h)) * SS;

    // ---- Phase 1: scores -> LDS (each wave owns 12 column tiles = 192 cols)
    v8f sacc[12] = {};
    #pragma unroll
    for (int kb = 0; kb < DHD; kb += 32) {
        const _Float16* qrow =
            q + (headBase + c * WW + qt * 16 + l15) * DHD + kb + o0;
        v8h lo = *(const v8h*)(qrow);
        v8h hi = *(const v8h*)(qrow + 8);
        v16h a = __builtin_shufflevector(lo, hi, 0, 1, 2, 3, 4, 5, 6, 7,
                                         8, 9, 10, 11, 12, 13, 14, 15);
        #pragma unroll
        for (int j = 0; j < 12; ++j) {
            int jcol = (wave * 12 + j) * 16 + l15;
            int kpos = (c - 1) * WW + jcol;
            v16h bf = {};
            if (kpos >= 0 && kpos < SS) {   // uniform per 16-wide tile
                bf = *(const v16h*)(k + (headBase + kpos) * DHD + kb +
                                    (hi16 ? 16 : 0));
            }
            sacc[j] = wmma_f16(a, bf, sacc[j]);
        }
    }

    const int mOff = hi16 ? 8 : 0;
    #pragma unroll
    for (int j = 0; j < 12; ++j) {
        int jcol = (wave * 12 + j) * 16 + l15;
        int kpos = (c - 1) * WW + jcol;
        bool inr = (kpos >= 0) && (kpos < SS);
        bool kok = inr && (mask[b * SS + (inr ? kpos : 0)] > 0);
        #pragma unroll
        for (int r = 0; r < 8; ++r) {
            int m = mOff + r;
            int ql = qt * 16 + m;
            int rel = jcol - WW - ql;
            bool ok = kok && (rel >= -WW) && (rel <= WW);
            sS[m * (3 * WW) + jcol] = ok ? sacc[j][r] : -1.0e9f;
        }
    }
    __syncthreads();

    // ---- Softmax: 8 lanes per row (rows stay inside one wave)
    {
        int row = tid >> 3, sub = tid & 7;
        float mx = -3.0e38f;
        for (int jj = sub; jj < 3 * WW; jj += 8)
            mx = fmaxf(mx, sS[row * (3 * WW) + jj]);
        #pragma unroll
        for (int d = 1; d < 8; d <<= 1) mx = fmaxf(mx, __shfl_xor(mx, d, 8));
        float sum = 0.0f;
        for (int jj = sub; jj < 3 * WW; jj += 8) {
            float e = __expf(sS[row * (3 * WW) + jj] - mx);
            sS[row * (3 * WW) + jj] = e;
            sum += e;
        }
        #pragma unroll
        for (int d = 1; d < 8; d <<= 1) sum += __shfl_xor(sum, d, 8);
        float inv = 1.0f / sum;
        for (int jj = sub; jj < 3 * WW; jj += 8)
            sS[row * (3 * WW) + jj] *= inv;
    }
    __syncthreads();

    // ---- Phase 2: out = P x V   (wave -> one 16-wide dh tile)
    v8f oacc = {};
    int dh = wave * 16 + l15;
    const _Float16* vrow = v + (headBase + 0) * 0 +          // keep types tidy
                           (((size_t)(b * HH + h)) * DHD + dh) * SS;
    for (int kb = 0; kb < 3 * WW; kb += 32) {
        const float* prow = sS + l15 * (3 * WW) + kb + o0;
        v16h a;
        #pragma unroll
        for (int i = 0; i < 8; ++i) {
            a[i]     = (_Float16)prow[i];
            a[i + 8] = (_Float16)prow[i + 8];
        }
        int kk0 = kb + (hi16 ? 16 : 0);
        int kpos0 = (c - 1) * WW + kk0;          // 16-aligned run of keys
        v16h bf = {};
        if (kpos0 >= 0 && kpos0 + 16 <= SS) {    // p == 0 outside -> zeros OK
            bf = *(const v16h*)(vrow + kpos0);
        }
        oacc = wmma_f16(a, bf, oacc);
    }
    #pragma unroll
    for (int r = 0; r < 8; ++r) {
        int m = mOff + r;
        size_t srow = (size_t)b * SS + c * WW + qt * 16 + m;
        aOut[srow * DD + h * DHD + wave * 16 + l15] = (_Float16)oacc[r];
    }
}

// ---------------------------------------------------------------------------
// Embedding + LayerNorm (one block per token)
// ---------------------------------------------------------------------------
__global__ __launch_bounds__(256) void embed_ln_kernel(
    const int* __restrict__ ids, const float* __restrict__ we,
    const float* __restrict__ pe, const float* __restrict__ te,
    const float* __restrict__ g, const float* __restrict__ bta,
    float* __restrict__ outF, _Float16* __restrict__ outH)
{
    __shared__ float red[256];
    int tok = blockIdx.x;
    int s = tok & (SS - 1);
    int id = ids[tok];
    int tid = threadIdx.x;

    float vv[3];
    #pragma unroll
    for (int i = 0; i < 3; ++i) {
        int d = tid + i * 256;
        vv[i] = we[(size_t)id * DD + d] + pe[(size_t)(s + 2) * DD + d] + te[d];
    }
    float su = block_sum_256(vv[0] + vv[1] + vv[2], red);
    float mu = su * (1.0f / DD);
    float qsum = 0.0f;
    #pragma unroll
    for (int i = 0; i < 3; ++i) { float d0 = vv[i] - mu; qsum += d0 * d0; }
    qsum = block_sum_256(qsum, red);
    float rstd = rsqrtf(qsum * (1.0f / DD) + 1e-5f);
    #pragma unroll
    for (int i = 0; i < 3; ++i) {
        int d = tid + i * 256;
        float o = (vv[i] - mu) * rstd * g[d] + bta[d];
        outF[(size_t)tok * DD + d] = o;
        outH[(size_t)tok * DD + d] = (_Float16)o;
    }
}

// LayerNorm over pre-summed input (residual already added by GEMM epilogue)
__global__ __launch_bounds__(256) void ln_kernel(
    const float* __restrict__ in, const float* __restrict__ g,
    const float* __restrict__ bta, float* __restrict__ outF,
    _Float16* __restrict__ outH)
{
    __shared__ float red[256];
    size_t row = blockIdx.x;
    int tid = threadIdx.x;
    const float* p = in + row * DD;
    float vv[3];
    #pragma unroll
    for (int i = 0; i < 3; ++i) vv[i] = p[tid + i * 256];
    float su = block_sum_256(vv[0] + vv[1] + vv[2], red);
    float mu = su * (1.0f / DD);
    float qsum = 0.0f;
    #pragma unroll
    for (int i = 0; i < 3; ++i) { float d0 = vv[i] - mu; qsum += d0 * d0; }
    qsum = block_sum_256(qsum, red);
    float rstd = rsqrtf(qsum * (1.0f / DD) + 1e-5f);
    #pragma unroll
    for (int i = 0; i < 3; ++i) {
        int d = tid + i * 256;
        float o = (vv[i] - mu) * rstd * g[d] + bta[d];
        outF[row * DD + d] = o;
        outH[row * DD + d] = (_Float16)o;
    }
}

// ---------------------------------------------------------------------------
// Classifier head
// ---------------------------------------------------------------------------
__global__ __launch_bounds__(256) void head_fc1_kernel(
    const float* __restrict__ x, const float* __restrict__ fc1w,
    const float* __restrict__ fc1b, float* __restrict__ p1)
{
    __shared__ float red[256];
    int n = blockIdx.x;
    for (int b = 0; b < BB; ++b) {
        const float* cls = x + (size_t)b * SS * DD;   // token 0 of sample b
        float s = 0.0f;
        for (int d = threadIdx.x; d < DD; d += 256)
            s += cls[d] * fc1w[(size_t)d * DD + n];
        s = block_sum_256(s, red);
        if (threadIdx.x == 0) p1[b * DD + n] = s + fc1b[n];
    }
}

__global__ __launch_bounds__(256) void head_final_kernel(
    const float* __restrict__ p1, const float* __restrict__ fc2w,
    const float* __restrict__ fc2b, const int* __restrict__ labels,
    float* __restrict__ out)
{
    __shared__ float red[256];
    float pred[BB];
    for (int b = 0; b < BB; ++b) {
        float s = 0.0f;
        for (int d = threadIdx.x; d < DD; d += 256)
            s += p1[b * DD + d] * fc2w[d];
        s = block_sum_256(s, red);
        pred[b] = s + fc2b[0];
    }
    if (threadIdx.x == 0) {
        float loss = 0.0f;
        for (int b = 0; b < BB; ++b) {
            float p = pred[b];
            float y = (float)labels[b];
            float lae = fmaxf(p, 0.0f) + log1pf(__expf(-fabsf(p)));
            loss += lae - p * y;
            out[1 + b] = 1.0f - p;
        }
        out[0] = loss / (float)BB;
    }
}

// ---------------------------------------------------------------------------
// Host orchestration
// ---------------------------------------------------------------------------
extern "C" void kernel_launch(void* const* d_in, const int* in_sizes, int n_in,
                              void* d_out, int out_size, void* d_ws, size_t ws_size,
                              hipStream_t stream)
{
    (void)in_sizes; (void)n_in; (void)out_size; (void)ws_size;

    // Inputs (setup_inputs() dict order, params flattened in insertion order)
    const int*   ids     = (const int*)  d_in[0];
    // d_in[1] = batch_segids (unused: single segment type)
    const int*   mask    = (const int*)  d_in[2];
    const int*   labels  = (const int*)  d_in[3];
    const float* word_e  = (const float*)d_in[4];
    const float* pos_e   = (const float*)d_in[5];
    const float* type_e  = (const float*)d_in[6];
    const float* emb_g   = (const float*)d_in[7];
    const float* emb_b   = (const float*)d_in[8];
    const float* wq      = (const float*)d_in[9];
    const float* bq      = (const float*)d_in[10];
    const float* wk      = (const float*)d_in[11];
    const float* bk      = (const float*)d_in[12];
    const float* wv      = (const float*)d_in[13];
    const float* bv      = (const float*)d_in[14];
    const float* wo      = (const float*)d_in[15];
    const float* bo      = (const float*)d_in[16];
    const float* ln1g    = (const float*)d_in[17];
    const float* ln1b    = (const float*)d_in[18];
    const float* w1      = (const float*)d_in[19];
    const float* b1      = (const float*)d_in[20];
    const float* w2      = (const float*)d_in[21];
    const float* b2      = (const float*)d_in[22];
    const float* ln2g    = (const float*)d_in[23];
    const float* ln2b    = (const float*)d_in[24];
    const float* fc1w    = (const float*)d_in[25];
    const float* fc1b    = (const float*)d_in[26];
    const float* fc2w    = (const float*)d_in[27];
    const float* fc2b    = (const float*)d_in[28];

    // Workspace carve-up
    char* wsp = (char*)d_ws;
    auto alloc = [&](size_t bytes) -> void* {
        void* p = (void*)wsp;
        wsp += (bytes + 255) & ~(size_t)255;
        return p;
    };
    float*    x     = (float*)   alloc((size_t)TOK * DD * 4);   // fp32 stream
    _Float16* xh    = (_Float16*)alloc((size_t)TOK * DD * 2);   // f16 GEMM A
    _Float16* qbuf  = (_Float16*)alloc((size_t)TOK * DD * 2);   // [B,H,S,DH]
    _Float16* kbuf  = (_Float16*)alloc((size_t)TOK * DD * 2);   // [B,H,S,DH]
    _Float16* vbuf  = (_Float16*)alloc((size_t)TOK * DD * 2);   // [B,H,DH,S]
    _Float16* abuf  = (_Float16*)alloc((size_t)TOK * DD * 2);   // attn out
    float*    tmp   = (float*)   alloc((size_t)TOK * DD * 4);   // pre-LN
    _Float16* hbuf  = (_Float16*)alloc((size_t)TOK * FF * 2);   // GELU(FFN1)
    _Float16* wpack = (_Float16*)alloc((size_t)DD * FF * 2);    // packed weight
    float*    p1    = (float*)   alloc((size_t)BB * DD * 4);

    const dim3 blk256(256), blk128(128);
    const dim3 gGemmD(TOK / 128, DD / 64);   // N = 768
    const dim3 gGemmF(TOK / 128, FF / 64);   // N = 3072
    const int  gPackDD = (DD * DD) / 256;
    const int  gPackDF = (DD * FF) / 256;
    const float qscale = 0.125f;             // 1/sqrt(64)

    // Embedding + LN
    embed_ln_kernel<<<TOK, blk256, 0, stream>>>(ids, word_e, pos_e, type_e,
                                                emb_g, emb_b, x, xh);

    for (int l = 0; l < LL; ++l) {
        const float* wq_l = wq + (size_t)l * DD * DD;
        const float* wk_l = wk + (size_t)l * DD * DD;
        const float* wv_l = wv + (size_t)l * DD * DD;
        const float* wo_l = wo + (size_t)l * DD * DD;
        const float* w1_l = w1 + (size_t)l * DD * FF;
        const float* w2_l = w2 + (size_t)l * FF * DD;

        // Q, K projections -> [B,H,S,DH] f16 (q pre-scaled); V -> [B,H,DH,S]
        pack_w_kernel<<<gPackDD, blk256, 0, stream>>>(wq_l, wpack, DD, DD);
        gemm_wmma_kernel<0><<<gGemmD, blk256, 0, stream>>>(
            xh, wpack, bq + l * DD, nullptr, qbuf, nullptr, DD, DD, qscale);
        pack_w_kernel<<<gPackDD, blk256, 0, stream>>>(wk_l, wpack, DD, DD);
        gemm_wmma_kernel<0><<<gGemmD, blk256, 0, stream>>>(
            xh, wpack, bk + l * DD, nullptr, kbuf, nullptr, DD, DD, 1.0f);
        pack_w_kernel<<<gPackDD, blk256, 0, stream>>>(wv_l, wpack, DD, DD);
        gemm_wmma_kernel<3><<<gGemmD, blk256, 0, stream>>>(
            xh, wpack, bv + l * DD, nullptr, vbuf, nullptr, DD, DD, 1.0f);

        // Sliding-window attention
        attn_kernel<<<BB * HH * CC * (WW / 16), blk128, 0, stream>>>(
            qbuf, kbuf, vbuf, mask, abuf);

        // Output projection + residual, then LN1
        pack_w_kernel<<<gPackDD, blk256, 0, stream>>>(wo_l, wpack, DD, DD);
        gemm_wmma_kernel<1><<<gGemmD, blk256, 0, stream>>>(
            abuf, wpack, bo + l * DD, tmp, nullptr, x, DD, DD, 1.0f);
        ln_kernel<<<TOK, blk256, 0, stream>>>(tmp, ln1g + l * DD, ln1b + l * DD,
                                              x, xh);

        // FFN1 (GELU) -> f16
        pack_w_kernel<<<gPackDF, blk256, 0, stream>>>(w1_l, wpack, DD, FF);
        gemm_wmma_kernel<2><<<gGemmF, blk256, 0, stream>>>(
            xh, wpack, b1 + l * FF, nullptr, hbuf, nullptr, FF, DD, 1.0f);

        // FFN2 + residual, then LN2
        pack_w_kernel<<<gPackDF, blk256, 0, stream>>>(w2_l, wpack, FF, DD);
        gemm_wmma_kernel<1><<<gGemmD, blk256, 0, stream>>>(
            hbuf, wpack, b2 + l * DD, tmp, nullptr, x, DD, FF, 1.0f);
        ln_kernel<<<TOK, blk256, 0, stream>>>(tmp, ln2g + l * DD, ln2b + l * DD,
                                              x, xh);
    }

    // Classifier head on CLS tokens
    head_fc1_kernel<<<DD, blk256, 0, stream>>>(x, fc1w, fc1b, p1);
    head_final_kernel<<<1, blk256, 0, stream>>>(p1, fc2w, fc2b, labels,
                                                (float*)d_out);
}
